// KeyCompressor_55645596287159
// MI455X (gfx1250) — matrix operand order repaired and checked by reading, not compiled, not run on hardware
//
#include <hip/hip_runtime.h>
#include <hip/hip_bf16.h>
#include <math.h>
#include <stdint.h>

typedef __bf16 bf16_t;
typedef __attribute__((ext_vector_type(16))) __bf16 v16bf;
typedef __attribute__((ext_vector_type(8)))  __bf16 v8bf;
typedef __attribute__((ext_vector_type(8)))  float  v8f;

#define FEAT 1024
#define GS   128
#define G    8
#define R    21
#define KH   8
#define KCB  64            // codebook size
#define DI   4096
#define DE   (G * R * KCB) // 10752
#define BTOK 4096          // B*S
#define NSLOT (G * R)      // 168
#define EPS_   1e-6f
#define LNEPS_ 1e-5f

#define KC 128             // GEMM2 LDS K-chunk (bf16 elements)

// s_wait_asynccnt via SOPP inline asm (ISA 01_flow_sync, opcode 74)
#define WAIT_ASYNC_4() asm volatile("s_wait_asynccnt 0x4" ::: "memory")
#define WAIT_ASYNC_0() asm volatile("s_wait_asynccnt 0x0" ::: "memory")

union Frag { v16bf v; v8bf h[2]; };

// A-matrix 16x32 bf16 fragment (row-major A, leading dim ld).
// Lane l<16: row=l, elems 0..7 = K k0+0..7, elems 8..15 = K k0+16..23
// Lane l>=16: row=l-16, elems 0..7 = K k0+8..15, elems 8..15 = K k0+24..31
__device__ inline v16bf load_a(const bf16_t* __restrict__ base, int ld, int row0, int k0, int lane) {
  const int r   = row0 + (lane & 15);
  const int klo = (lane & 16) >> 1;            // 0 or 8
  const bf16_t* p = base + (size_t)r * ld + k0 + klo;
  Frag f;
  f.h[0] = *reinterpret_cast<const v8bf*>(p);
  f.h[1] = *reinterpret_cast<const v8bf*>(p + 16);
  return f.v;
}

// B-matrix 32x16 bf16 fragment; source stored row-major [N, K] (weights or LDS
// tile). Lane l<16: col=l, K=k0+0..7,k0+16..23; lane l>=16: col=l-16, K=k0+8..15,k0+24..31.
__device__ inline v16bf load_b(const bf16_t* __restrict__ base, int ld, int n0, int k0, int lane) {
  const int c  = n0 + (lane & 15);
  const int kh = (lane & 16) >> 1;             // 0 or 8
  const bf16_t* p = base + (size_t)c * ld + k0 + kh;
  Frag f;
  f.h[0] = *reinterpret_cast<const v8bf*>(p);
  f.h[1] = *reinterpret_cast<const v8bf*>(p + 16);
  return f.v;
}

// Async-copy one 64col x KC B-chunk (row-major [64][KC]) into LDS.
// 64*KC/8 = 1024 16-byte units; 256 threads x 4 units each.
__device__ inline void async_copy_B(const bf16_t* __restrict__ gsrc, int ldg,
                                    bf16_t* sdst, int tid) {
#pragma unroll
  for (int i = 0; i < 4; ++i) {
    const int j = tid + i * 256;
    const int c = j >> 4;        // column 0..63
    const int u = j & 15;        // 16B unit within column
    const bf16_t* gp = gsrc + (size_t)c * ldg + u * 8;
    bf16_t* lp = sdst + c * KC + u * 8;
    const uint32_t lds_off = (uint32_t)(uintptr_t)lp;   // low 32 bits = LDS offset
    const uint64_t gaddr   = (uint64_t)(uintptr_t)gp;
    asm volatile("global_load_async_to_lds_b128 %0, %1, off"
                 :: "v"(lds_off), "v"(gaddr) : "memory");
  }
}

// ---------------------------------------------------------------------------
// 1) Fused scale-div + L2 norm + LayerNorm -> bf16 xn, f32 prescale
// ---------------------------------------------------------------------------
__global__ __launch_bounds__(256) void prep_kernel(
    const float* __restrict__ x, const float* __restrict__ ls,
    const float* __restrict__ lnw, const float* __restrict__ lnb,
    bf16_t* __restrict__ xn, float* __restrict__ prescale_out) {
  const int t   = blockIdx.x;
  const int tid = threadIdx.x;
  const float* xr = x + (size_t)t * FEAT;

  float y[4];
  float s1 = 0.f, s2 = 0.f;
#pragma unroll
  for (int i = 0; i < 4; ++i) {
    const int f = tid + i * 256;
    y[i] = xr[f] / (ls[f] + EPS_);
    s1 += y[i];
    s2 += y[i] * y[i];
  }
#pragma unroll
  for (int m = 16; m >= 1; m >>= 1) {
    s1 += __shfl_xor(s1, m, 32);
    s2 += __shfl_xor(s2, m, 32);
  }
  __shared__ float red1[8], red2[8];
  const int wv = tid >> 5, ln = tid & 31;
  if (ln == 0) { red1[wv] = s1; red2[wv] = s2; }
  __syncthreads();
  float t1 = 0.f, t2 = 0.f;
#pragma unroll
  for (int w = 0; w < 8; ++w) { t1 += red1[w]; t2 += red2[w]; }

  const float pres = sqrtf(t2);
  const float inv  = 1.f / (pres + EPS_);
  const float mu   = t1 * inv * (1.f / FEAT);
  const float var  = t2 * inv * inv * (1.f / FEAT) - mu * mu;
  const float rstd = rsqrtf(var + LNEPS_);

#pragma unroll
  for (int i = 0; i < 4; ++i) {
    const int f = tid + i * 256;
    const float z = y[i] * inv;
    const float v = (z - mu) * rstd * lnw[f] + lnb[f];
    xn[(size_t)t * FEAT + f] = (bf16_t)v;
  }
  if (tid == 0) prescale_out[t] = pres;
}

// ---------------------------------------------------------------------------
// 2) f32 -> bf16 conversion (weights)
// ---------------------------------------------------------------------------
__global__ __launch_bounds__(256) void cvt_bf16_kernel(
    const float* __restrict__ src, bf16_t* __restrict__ dst, size_t n) {
  size_t i = (size_t)blockIdx.x * blockDim.x + threadIdx.x;
  const size_t stride = (size_t)gridDim.x * blockDim.x;
  for (; i < n; i += stride) dst[i] = (bf16_t)src[i];
}

// ---------------------------------------------------------------------------
// 3) Codebook centers (T-matrix is +/-1 sparse -> elementwise)
//    centers[g,r,k=A*8+B, 2d]   = p[g,r,A,d,0] - p[g,r,B,d,1]
//    centers[g,r,k=A*8+B, 2d+1] = p[g,r,B,d,0] + p[g,r,A,d,1]
// ---------------------------------------------------------------------------
__global__ __launch_bounds__(256) void centers_kernel(
    const float* __restrict__ params, float* __restrict__ centers) {
  const int i = blockIdx.x * 256 + threadIdx.x;
  if (i >= NSLOT * KCB * (GS / 2)) return;
  const int d  = i & 63;
  const int k  = (i >> 6) & 63;
  const int gr = i >> 12;
  const int A  = k >> 3, Bc = k & 7;
  const float* pb = params + (size_t)gr * (KH * (GS / 2) * 2);
  const float a0 = pb[(A * 64 + d) * 2 + 0];
  const float a1 = pb[(A * 64 + d) * 2 + 1];
  const float b0 = pb[(Bc * 64 + d) * 2 + 0];
  const float b1 = pb[(Bc * 64 + d) * 2 + 1];
  float* cb = centers + ((size_t)gr * KCB + k) * GS;
  cb[2 * d]     = a0 - b1;
  cb[2 * d + 1] = b0 + a1;
}

// ---------------------------------------------------------------------------
// 4) GEMM1: h = GELU(xn @ W1^T + b1), bf16 out.
//    Block = 8 waves (2M x 4N); wave = 32M x 64N; 8 WMMAs / K-step.
//    Grid: (DI/256, BTOK/64)
// ---------------------------------------------------------------------------
__global__ __launch_bounds__(256) void gemm1_wmma(
    const bf16_t* __restrict__ A, const bf16_t* __restrict__ W,
    const float* __restrict__ bias, bf16_t* __restrict__ H) {
  const int lane = threadIdx.x & 31;
  const int wave = threadIdx.x >> 5;
  const int row0 = blockIdx.y * 64 + (wave & 1) * 32;
  const int n0   = blockIdx.x * 256 + (wave >> 1) * 64;

  v8f acc[2][4] = {};
  for (int k0 = 0; k0 < FEAT; k0 += 32) {
    const v16bf a0 = load_a(A, FEAT, row0, k0, lane);
    const v16bf a1 = load_a(A, FEAT, row0 + 16, k0, lane);
#pragma unroll
    for (int t = 0; t < 4; ++t) {
      const v16bf b = load_b(W, FEAT, n0 + t * 16, k0, lane);
      acc[0][t] = __builtin_amdgcn_wmma_f32_16x16x32_bf16(false, a0, false, b, (short)0, acc[0][t], false, false);
      acc[1][t] = __builtin_amdgcn_wmma_f32_16x16x32_bf16(false, a1, false, b, (short)0, acc[1][t], false, false);
    }
  }

  const int half = lane >> 4;
  const int cIn  = lane & 15;
#pragma unroll
  for (int t = 0; t < 4; ++t) {
    const int col = n0 + t * 16 + cIn;
    const float bv = bias[col];
#pragma unroll
    for (int mi = 0; mi < 2; ++mi) {
#pragma unroll
      for (int v = 0; v < 8; ++v) {
        const int r = row0 + mi * 16 + v + 8 * half;
        const float xv = acc[mi][t][v] + bv;
        const float gl = 0.5f * xv * (1.0f + erff(xv * 0.70710678118f));
        H[(size_t)r * DI + col] = (bf16_t)gl;
      }
    }
  }
}

// ---------------------------------------------------------------------------
// 5) GEMM2 + fused bias + argmax over the 64 logits of one (g,r) slot.
//    Wave = 32M x 64N (one slot); grid (NSLOT, BTOK/256); logits never stored.
//    B tile staged in LDS via async-to-LDS, double-buffered (8x reuse across
//    the block's waves); A fragments stream from global (L2-resident).
// ---------------------------------------------------------------------------
__global__ __launch_bounds__(256) void gemm2_wmma(
    const bf16_t* __restrict__ Hm, const bf16_t* __restrict__ W,
    const float* __restrict__ bias, int* __restrict__ idx_out) {
  const int tid  = threadIdx.x;
  const int lane = tid & 31;
  const int wave = tid >> 5;
  const int slot = blockIdx.x;                 // 0..167 = g*R + r
  const int row0 = blockIdx.y * 256 + wave * 32;
  const int n0   = slot * KCB;

  __shared__ bf16_t sB[2][KCB * KC];           // 2 x 16 KB double buffer

  const bf16_t* wbase = W + (size_t)n0 * DI;

  v8f acc[2][4] = {};

  // prologue: stage chunk 0
  async_copy_B(wbase, DI, &sB[0][0], tid);

  const int NCHUNK = DI / KC;                  // 32
  for (int ch = 0; ch < NCHUNK; ++ch) {
    const int kc = ch * KC;
    if (ch + 1 < NCHUNK) {
      async_copy_B(wbase + (kc + KC), DI, &sB[(ch + 1) & 1][0], tid);
      WAIT_ASYNC_4();                          // chunk ch landed (in-order)
    } else {
      WAIT_ASYNC_0();
    }
    __syncthreads();                           // all waves' chunks visible

    const bf16_t* sb = &sB[ch & 1][0];
#pragma unroll
    for (int kk = 0; kk < KC; kk += 32) {
      const v16bf a0 = load_a(Hm, DI, row0, kc + kk, lane);
      const v16bf a1 = load_a(Hm, DI, row0 + 16, kc + kk, lane);
#pragma unroll
      for (int t = 0; t < 4; ++t) {
        const v16bf b = load_b(sb, KC, t * 16, kk, lane);
        acc[0][t] = __builtin_amdgcn_wmma_f32_16x16x32_bf16(false, a0, false, b, (short)0, acc[0][t], false, false);
        acc[1][t] = __builtin_amdgcn_wmma_f32_16x16x32_bf16(false, a1, false, b, (short)0, acc[1][t], false, false);
      }
    }
    __syncthreads();                           // done reading before rewrite
  }

  const int half = lane >> 4;
  const int cIn  = lane & 15;
  float bv[4];
#pragma unroll
  for (int t = 0; t < 4; ++t) bv[t] = bias[n0 + t * 16 + cIn];

#pragma unroll
  for (int mi = 0; mi < 2; ++mi) {
#pragma unroll
    for (int v = 0; v < 8; ++v) {
      float best = acc[mi][0][v] + bv[0];
      int   bi   = cIn;
#pragma unroll
      for (int t = 1; t < 4; ++t) {
        const float val = acc[mi][t][v] + bv[t];
        if (val > best) { best = val; bi = t * 16 + cIn; }   // '>' keeps first max
      }
#pragma unroll
      for (int m = 1; m < 16; m <<= 1) {       // masks 1..8 stay within 16-lane half
        const float ov = __shfl_xor(best, m, 32);
        const int   oi = __shfl_xor(bi, m, 32);
        if (ov > best || (ov == best && oi < bi)) { best = ov; bi = oi; }
      }
      if (cIn == 0) {
        const int r = row0 + mi * 16 + v + 8 * half;
        idx_out[(size_t)r * NSLOT + slot] = bi;
      }
    }
  }
}

// ---------------------------------------------------------------------------
// 6) Decode: gather centers over R residuals, rescale, write quant (+loss=0)
// ---------------------------------------------------------------------------
__global__ __launch_bounds__(256) void decode_kernel(
    const float* __restrict__ centers, const int* __restrict__ idx,
    const float* __restrict__ prescale, const float* __restrict__ ls,
    float* __restrict__ out) {
  const int t   = blockIdx.x;
  const int tid = threadIdx.x;
  const float ps = prescale[t] + EPS_;
#pragma unroll
  for (int i = 0; i < 4; ++i) {
    const int f = tid + i * 256;
    const int g = f >> 7;
    const int d = f & 127;
    const int* ip = idx + (size_t)t * NSLOT + g * R;
    const float* cg = centers + (size_t)g * R * KCB * GS;
    float acc = 0.f;
#pragma unroll
    for (int r = 0; r < R; ++r)
      acc += cg[((size_t)r * KCB + ip[r]) * GS + d];
    out[(size_t)t * FEAT + f] = acc * ps * (ls[f] + EPS_);
  }
  if (t == 0 && tid == 0)
    out[(size_t)BTOK * FEAT + BTOK] = 0.0f;   // commitment_loss
}

// ---------------------------------------------------------------------------
extern "C" void kernel_launch(void* const* d_in, const int* in_sizes, int n_in,
                              void* d_out, int out_size, void* d_ws, size_t ws_size,
                              hipStream_t stream) {
  const float* x   = (const float*)d_in[0];
  const float* ls  = (const float*)d_in[1];
  const float* lnw = (const float*)d_in[2];
  const float* lnb = (const float*)d_in[3];
  const float* W1  = (const float*)d_in[4];
  const float* b1  = (const float*)d_in[5];
  const float* W2  = (const float*)d_in[6];
  const float* b2  = (const float*)d_in[7];
  const float* cbp = (const float*)d_in[8];
  float* out = (float*)d_out;

  char* ws = (char*)d_ws;
  size_t off = 0;
  auto alloc = [&](size_t bytes) -> void* {
    void* p = ws + off;
    off = (off + bytes + 255) & ~(size_t)255;
    return p;
  };
  bf16_t* xn    = (bf16_t*)alloc((size_t)BTOK * FEAT * 2);
  bf16_t* w1b   = (bf16_t*)alloc((size_t)DI * FEAT * 2);
  bf16_t* w2b   = (bf16_t*)alloc((size_t)DE * DI * 2);
  bf16_t* hbuf  = (bf16_t*)alloc((size_t)BTOK * DI * 2);
  float*  cent  = (float*)alloc((size_t)NSLOT * KCB * GS * 4);
  int*    idxb  = (int*)alloc((size_t)BTOK * NSLOT * 4);

  float* prescale = out + (size_t)BTOK * FEAT;  // d_out prescale region

  prep_kernel<<<BTOK, 256, 0, stream>>>(x, ls, lnw, lnb, xn, prescale);
  cvt_bf16_kernel<<<2048, 256, 0, stream>>>(W1, w1b, (size_t)DI * FEAT);
  cvt_bf16_kernel<<<4096, 256, 0, stream>>>(W2, w2b, (size_t)DE * DI);
  centers_kernel<<<(NSLOT * KCB * (GS / 2) + 255) / 256, 256, 0, stream>>>(cbp, cent);
  gemm1_wmma<<<dim3(DI / 256, BTOK / 64), 256, 0, stream>>>(xn, w1b, b1, hbuf);
  gemm2_wmma<<<dim3(NSLOT, BTOK / 256), 256, 0, stream>>>(hbuf, w2b, b2, idxb);
  decode_kernel<<<BTOK, 256, 0, stream>>>(cent, idxb, prescale, ls, out);
}
